// MolecularGraphNeuralNetwork_3899830305296
// MI455X (gfx1250) — matrix operand order repaired and checked by reading, not compiled, not run on hardware
//
#include <hip/hip_runtime.h>
#include <hip/hip_bf16.h>
#include <math.h>

// ---------------------------------------------------------------------------
// MolecularGraphNeuralNetwork for MI455X (gfx1250, wave32, WMMA)
// bf16 activations + fragment-packed bf16 weights; f32 WMMA accumulation.
// ---------------------------------------------------------------------------

typedef __bf16 bf16_t;
typedef __attribute__((ext_vector_type(16))) __bf16 v16bf;
typedef __attribute__((ext_vector_type(8)))  __bf16 bf16x8;
typedef __attribute__((ext_vector_type(8)))  float  v8f;

#define DIM       512
#define N_MOL     128
#define APM       32          // atoms per molecule
#define N_ATOMS   4096
#define NHID      6
#define NOUT      3

__device__ __forceinline__ v8f wmma_bf16(v16bf a, v16bf b, v8f c) {
  // D = A(16x32 bf16) * B(32x16 bf16) + C(16x16 f32)
  return __builtin_amdgcn_wmma_f32_16x16x32_bf16(
      /*neg_a=*/false, a, /*neg_b=*/false, b,
      /*c_mod=*/(short)0, c, /*reuse_a=*/false, /*reuse_b=*/false);
}

// A fragment: bf16 row-major [M,K]; lane<16 row=lane K+0..15,
// lane>=16 row=lane-16 K+16..31. Two contiguous 16B loads.
__device__ __forceinline__ v16bf ld_afrag(const bf16_t* __restrict__ A, int K,
                                          int rowbase, int kb, int lane) {
  int r    = rowbase + (lane & 15);
  int koff = kb + ((lane >> 4) << 4);
  const bf16_t* p = A + (size_t)r * K + koff;
  bf16x8 lo = *(const bf16x8*)p;         // K +0..7
  bf16x8 hi = *(const bf16x8*)(p + 8);   // K +8..15
  return __builtin_shufflevector(lo, hi, 0, 1, 2, 3, 4, 5, 6, 7,
                                 8, 9, 10, 11, 12, 13, 14, 15);
}

// B fragment from fragment-packed weights: 32 contiguous bytes per lane.
__device__ __forceinline__ v16bf ld_bfrag(const bf16_t* __restrict__ P, int N16,
                                          int ks, int tn, int lane) {
  const bf16_t* p = P + (((size_t)(ks * N16 + tn) * 32 + lane) << 4);
  bf16x8 lo = *(const bf16x8*)p;
  bf16x8 hi = *(const bf16x8*)(p + 8);
  return __builtin_shufflevector(lo, hi, 0, 1, 2, 3, 4, 5, 6, 7,
                                 8, 9, 10, 11, 12, 13, 14, 15);
}

// ---------------------------------------------------------------------------
// Pack f32 W[K,N] -> bf16 fragment-native layout:
//   P[((ks*(N/16)+tn)*32+lane)*16 + j] = W[ks*32 + kh + j][tn*16 + cl]
//   cl = lane&15, kh = (lane>>4)*16
// ---------------------------------------------------------------------------
__global__ void pack_weights(const float* __restrict__ W, bf16_t* __restrict__ P,
                             int K, int N) {
  size_t total = (size_t)K * N;
  int N16 = N >> 4;
  for (size_t idx = (size_t)blockIdx.x * blockDim.x + threadIdx.x; idx < total;
       idx += (size_t)gridDim.x * blockDim.x) {
    int j      = (int)(idx & 15);
    size_t t   = idx >> 4;
    int lane   = (int)(t & 31);
    t >>= 5;
    int tn     = (int)(t % N16);
    int ks     = (int)(t / N16);
    int cl     = lane & 15;
    int kh     = (lane >> 4) << 4;
    int row    = ks * 32 + kh + j;
    int col    = tn * 16 + cl;
    P[idx] = (bf16_t)W[(size_t)row * N + col];
  }
}

// ---------------------------------------------------------------------------
// Embedding gather -> bf16: v[atom,:] = bf16(emb[fp[atom],:])  (both graphs)
// ---------------------------------------------------------------------------
__global__ void embed_kernel(const int* __restrict__ fp1,
                             const int* __restrict__ fp2,
                             const float* __restrict__ emb,
                             bf16_t* __restrict__ v1,
                             bf16_t* __restrict__ v2) {
  int atom  = blockIdx.x;
  int which = blockIdx.y;
  const int* fp = which ? fp2 : fp1;
  bf16_t*    v  = which ? v2  : v1;
  int tok = fp[atom];
  const float* src = emb + (size_t)tok * DIM;
  bf16_t* dst = v + (size_t)atom * DIM;
  for (int d = threadIdx.x; d < DIM; d += blockDim.x) dst[d] = (bf16_t)src[d];
}

// ---------------------------------------------------------------------------
// C[M,N] = act(A[M,K] @ W[K,N] + bias[N]); A bf16 row-major, P packed bf16,
// C bf16. Macro tile 64x128, 8 waves (2x4), wave tile 32x32 (4 WMMA/K-step).
// Requires M%64==0, N%128==0, K%32==0.
// ---------------------------------------------------------------------------
__global__ __launch_bounds__(256) void wmma_gemm_bf16(
    const bf16_t* __restrict__ A, const bf16_t* __restrict__ P,
    const float* __restrict__ bias, bf16_t* __restrict__ C,
    int M, int N, int K, int do_relu) {
  const int tid     = threadIdx.x;
  const int wave    = tid >> 5;
  const int lane    = tid & 31;
  const int wm      = wave >> 2;                     // 0..1
  const int wn      = wave & 3;                      // 0..3
  const int rowbase = blockIdx.y * 64 + wm * 32;
  const int colbase = blockIdx.x * 128 + wn * 32;
  const int N16     = N >> 4;
  const int tn0     = colbase >> 4;

  v8f acc[2][2];
  #pragma unroll
  for (int mt = 0; mt < 2; ++mt)
    #pragma unroll
    for (int nt = 0; nt < 2; ++nt)
      acc[mt][nt] = (v8f){0.f, 0.f, 0.f, 0.f, 0.f, 0.f, 0.f, 0.f};

  for (int kb = 0; kb < K; kb += 32) {
    const int ks = kb >> 5;
    if (kb + 32 < K) {  // prefetch next K-step (global_prefetch_b8)
      __builtin_prefetch(A + (size_t)(rowbase + (lane & 15)) * K + kb + 32, 0, 1);
      __builtin_prefetch(P + (((size_t)((ks + 1) * N16 + tn0) * 32 + lane) << 4), 0, 1);
    }
    v16bf a0 = ld_afrag(A, K, rowbase,      kb, lane);
    v16bf a1 = ld_afrag(A, K, rowbase + 16, kb, lane);
    v16bf b0 = ld_bfrag(P, N16, ks, tn0,     lane);
    v16bf b1 = ld_bfrag(P, N16, ks, tn0 + 1, lane);
    acc[0][0] = wmma_bf16(a0, b0, acc[0][0]);
    acc[0][1] = wmma_bf16(a0, b1, acc[0][1]);
    acc[1][0] = wmma_bf16(a1, b0, acc[1][0]);
    acc[1][1] = wmma_bf16(a1, b1, acc[1][1]);
  }

  // Epilogue per ISA C/D layout: VGPR r -> row r + (lane<16?0:8), col = lane%16
  const int cl = lane & 15;
  const int ro = (lane >> 4) << 3;
  #pragma unroll
  for (int mt = 0; mt < 2; ++mt) {
    #pragma unroll
    for (int nt = 0; nt < 2; ++nt) {
      int col = colbase + nt * 16 + cl;
      int r0  = rowbase + mt * 16 + ro;
      float bv = bias[col];
      #pragma unroll
      for (int r = 0; r < 8; ++r) {
        float x = acc[mt][nt][r] + bv;
        if (do_relu) x = fmaxf(x, 0.0f);
        C[(size_t)(r0 + r) * N + col] = (bf16_t)x;
      }
    }
  }
}

// ---------------------------------------------------------------------------
// Per-molecule: g = h + A_blk @ h (A_blk = 32x32 diag block of adj, f32 0/1),
// then v = g / max(||g_row||, 1e-12). h,v bf16. One 4-wave WG per molecule.
// ---------------------------------------------------------------------------
__global__ __launch_bounds__(128) void block_adj_norm(
    const bf16_t* __restrict__ H, const float* __restrict__ Adj,
    bf16_t* __restrict__ V) {
  const int m    = blockIdx.x;
  const int tid  = threadIdx.x;
  const int wave = tid >> 5;
  const int lane = tid & 31;

  const bf16_t* hblk = H   + (size_t)m * APM * DIM;
  const float*  ablk = Adj + (size_t)(m * APM) * N_ATOMS + (size_t)m * APM;
  bf16_t*       vblk = V   + (size_t)m * APM * DIM;

  __shared__ float rowsq[APM];
  __shared__ float rinv[APM];
  if (tid < APM) rowsq[tid] = 0.0f;
  __syncthreads();

  // Adjacency A-fragments (K=32): rows 0..15 (lo), 16..31 (hi). 0/1 exact bf16.
  v16bf a_lo, a_hi;
  const int ar = lane & 15;
  const int ak = (lane >> 4) << 4;
  #pragma unroll
  for (int j = 0; j < 8; ++j) {
    a_lo[j]     = (bf16_t)ablk[(size_t)ar * N_ATOMS + ak + j];
    a_lo[8 + j] = (bf16_t)ablk[(size_t)ar * N_ATOMS + ak + 8 + j];
    a_hi[j]     = (bf16_t)ablk[(size_t)(ar + 16) * N_ATOMS + ak + j];
    a_hi[8 + j] = (bf16_t)ablk[(size_t)(ar + 16) * N_ATOMS + ak + 8 + j];
  }

  const int cl = lane & 15;
  const int kh = (lane >> 4) << 4;   // B fragment K half
  const int ro = (lane >> 4) << 3;   // C fragment row offset

  float sq_lo[8] = {0, 0, 0, 0, 0, 0, 0, 0};
  float sq_hi[8] = {0, 0, 0, 0, 0, 0, 0, 0};

  for (int t = 0; t < 8; ++t) {      // 8 N-tiles of 16 in this wave's strip
    const int col = wave * 128 + t * 16 + cl;

    v16bf bfrag;                     // h as B (32x16 slice), already bf16
    #pragma unroll
    for (int j = 0; j < 16; ++j)
      bfrag[j] = hblk[(size_t)(kh + j) * DIM + col];

    v8f glo, ghi;                    // seed with h (g = h + A@h)
    #pragma unroll
    for (int r = 0; r < 8; ++r) {
      glo[r] = (float)hblk[(size_t)(ro + r) * DIM + col];
      ghi[r] = (float)hblk[(size_t)(16 + ro + r) * DIM + col];
    }
    glo = wmma_bf16(a_lo, bfrag, glo);
    ghi = wmma_bf16(a_hi, bfrag, ghi);

    #pragma unroll
    for (int r = 0; r < 8; ++r) {
      vblk[(size_t)(ro + r) * DIM + col]      = (bf16_t)glo[r];
      vblk[(size_t)(16 + ro + r) * DIM + col] = (bf16_t)ghi[r];
      sq_lo[r] += glo[r] * glo[r];
      sq_hi[r] += ghi[r] * ghi[r];
    }
  }

  #pragma unroll
  for (int r = 0; r < 8; ++r) {
    atomicAdd(&rowsq[ro + r],      sq_lo[r]);
    atomicAdd(&rowsq[16 + ro + r], sq_hi[r]);
  }
  __threadfence_block();
  __syncthreads();
  if (tid < APM) rinv[tid] = 1.0f / fmaxf(sqrtf(rowsq[tid]), 1e-12f);
  __syncthreads();

  for (int idx = tid; idx < APM * DIM; idx += 128) {
    int row = idx >> 9;
    vblk[idx] = (bf16_t)((float)vblk[idx] * rinv[row]);
  }
}

// ---------------------------------------------------------------------------
// Segment-sum pooling (f32 accumulate, bf16 out), concatenated [128, 1024].
// ---------------------------------------------------------------------------
__global__ void pool_kernel(const bf16_t* __restrict__ v1,
                            const bf16_t* __restrict__ v2,
                            bf16_t* __restrict__ mv) {
  int m = blockIdx.x;
  for (int c = threadIdx.x; c < DIM; c += blockDim.x) {
    float s1 = 0.f, s2 = 0.f;
    for (int i = 0; i < APM; ++i) {
      s1 += (float)v1[((size_t)m * APM + i) * DIM + c];
      s2 += (float)v2[((size_t)m * APM + i) * DIM + c];
    }
    mv[(size_t)m * (2 * DIM) + c]       = (bf16_t)s1;
    mv[(size_t)m * (2 * DIM) + DIM + c] = (bf16_t)s2;
  }
}

// ---------------------------------------------------------------------------
// out[m] = hb[m,:] . Wprop[:,0] + bprop  (f32 out)
// ---------------------------------------------------------------------------
__global__ __launch_bounds__(128) void final_dot(
    const bf16_t* __restrict__ hb, const float* __restrict__ Wprop,
    const float* __restrict__ bprop, float* __restrict__ out) {
  int m = blockIdx.x;
  __shared__ float red[128];
  float s = 0.f;
  for (int d = threadIdx.x; d < DIM; d += 128)
    s += (float)hb[(size_t)m * DIM + d] * Wprop[d];
  red[threadIdx.x] = s;
  __syncthreads();
  for (int off = 64; off > 0; off >>= 1) {
    if (threadIdx.x < off) red[threadIdx.x] += red[threadIdx.x + off];
    __syncthreads();
  }
  if (threadIdx.x == 0) out[m] = red[0] + bprop[0];
}

// ---------------------------------------------------------------------------
extern "C" void kernel_launch(void* const* d_in, const int* in_sizes, int n_in,
                              void* d_out, int out_size, void* d_ws, size_t ws_size,
                              hipStream_t stream) {
  const int*   fp1   = (const int*)  d_in[0];
  const int*   fp2   = (const int*)  d_in[1];
  const float* adj1  = (const float*)d_in[2];
  const float* adj2  = (const float*)d_in[3];
  /* d_in[4] = seg — fixed layout (32 atoms/molecule), unused */
  const float* emb   = (const float*)d_in[5];
  const float* Wf    = (const float*)d_in[6];   // [6,512,512]
  const float* bf    = (const float*)d_in[7];   // [6,512]
  const float* Wmlp  = (const float*)d_in[8];   // [1024,512]
  const float* bmlp  = (const float*)d_in[9];   // [512]
  const float* Wout  = (const float*)d_in[10];  // [3,512,512]
  const float* boutp = (const float*)d_in[11];  // [3,512]
  const float* Wprop = (const float*)d_in[12];  // [512,1]
  const float* bprop = (const float*)d_in[13];  // [1]
  float* out = (float*)d_out;                   // [128,1]

  // Workspace carve-up (bf16 activations + packed bf16 weights, ~18 MB)
  char* base = (char*)d_ws;
  bf16_t* v1  = (bf16_t*)base; base += (size_t)N_ATOMS * DIM * 2;
  bf16_t* v2  = (bf16_t*)base; base += (size_t)N_ATOMS * DIM * 2;
  bf16_t* h   = (bf16_t*)base; base += (size_t)N_ATOMS * DIM * 2;
  bf16_t* mv  = (bf16_t*)base; base += (size_t)N_MOL * 2 * DIM * 2;
  bf16_t* hb1 = (bf16_t*)base; base += (size_t)N_MOL * DIM * 2;
  bf16_t* hb2 = (bf16_t*)base; base += (size_t)N_MOL * DIM * 2;
  bf16_t* WfP   = (bf16_t*)base; base += (size_t)NHID * DIM * DIM * 2;
  bf16_t* WmlpP = (bf16_t*)base; base += (size_t)(2 * DIM) * DIM * 2;
  bf16_t* WoutP = (bf16_t*)base; base += (size_t)NOUT * DIM * DIM * 2;

  // 0) Pack all weights to fragment-native bf16 (once; L2-resident afterwards)
  for (int l = 0; l < NHID; ++l)
    pack_weights<<<256, 256, 0, stream>>>(Wf + (size_t)l * DIM * DIM,
                                          WfP + (size_t)l * DIM * DIM, DIM, DIM);
  pack_weights<<<256, 256, 0, stream>>>(Wmlp, WmlpP, 2 * DIM, DIM);
  for (int l = 0; l < NOUT; ++l)
    pack_weights<<<256, 256, 0, stream>>>(Wout + (size_t)l * DIM * DIM,
                                          WoutP + (size_t)l * DIM * DIM, DIM, DIM);

  // 1) Embedding gather (f32 table -> bf16 activations)
  embed_kernel<<<dim3(N_ATOMS, 2), 256, 0, stream>>>(fp1, fp2, emb, v1, v2);

  // 2) 6 GNN layers per graph: h = relu(v@Wf+bf); v = normalize(h + A_blk@h)
  for (int g = 0; g < 2; ++g) {
    bf16_t*      v   = g ? v2   : v1;
    const float* adj = g ? adj2 : adj1;
    for (int l = 0; l < NHID; ++l) {
      wmma_gemm_bf16<<<dim3(DIM / 128, N_ATOMS / 64), 256, 0, stream>>>(
          v, WfP + (size_t)l * DIM * DIM, bf + (size_t)l * DIM, h,
          N_ATOMS, DIM, DIM, /*relu=*/1);
      block_adj_norm<<<N_MOL, 128, 0, stream>>>(h, adj, v);
    }
  }

  // 3) Sum pooling into concatenated [128, 1024] bf16
  pool_kernel<<<N_MOL, 256, 0, stream>>>(v1, v2, mv);

  // 4) MLP head
  wmma_gemm_bf16<<<dim3(DIM / 128, N_MOL / 64), 256, 0, stream>>>(
      mv, WmlpP, bmlp, hb1, N_MOL, DIM, 2 * DIM, /*relu=*/1);
  bf16_t* cur = hb1;
  bf16_t* nxt = hb2;
  for (int l = 0; l < NOUT; ++l) {
    wmma_gemm_bf16<<<dim3(DIM / 128, N_MOL / 64), 256, 0, stream>>>(
        cur, WoutP + (size_t)l * DIM * DIM, boutp + (size_t)l * DIM, nxt,
        N_MOL, DIM, DIM, /*relu=*/1);
    bf16_t* t = cur; cur = nxt; nxt = t;
  }

  // 5) Final projection to [128,1] f32
  final_dot<<<N_MOL, 128, 0, stream>>>(cur, Wprop, bprop, out);
}
